// CrossAttentionBlock_62165356642512
// MI455X (gfx1250) — matrix operand order repaired and checked by reading, not compiled, not run on hardware
//
#include <hip/hip_runtime.h>

#define C_CH 128
#define BN_SC 0.9999950000374997f   // 1/sqrt(1+1e-5)

typedef __attribute__((ext_vector_type(16))) _Float16 v16h;
typedef __attribute__((ext_vector_type(8)))  _Float16 v8h;
typedef __attribute__((ext_vector_type(8)))  float    v8f;

static __device__ __forceinline__ v8f wmma16(v16h a, v16h b, v8f c) {
    // v_wmma_f32_16x16x32_f16: (neg_a, A, neg_b, B, c_mod, C, reuse_a, reuse_b)
    return __builtin_amdgcn_wmma_f32_16x16x32_f16(false, a, false, b, (short)0, c, false, false);
}

static __device__ __forceinline__ v16h frag_cat8(v8h a, v8h b) {
    v16h r;
#pragma unroll
    for (int i = 0; i < 8; ++i) { r[i] = a[i]; r[i + 8] = b[i]; }
    return r;
}

// A-matrix (16x32 f16) fragment for one 32-wide K chunk.
// Contiguous k order used here (same as B): lane holds row (lane&15); halfword e -> K = 16*half + e.
// => 16 contiguous halfwords at K offset 16*half.
static __device__ __forceinline__ v16h load_a_f16(const _Float16* row, int t, int half) {
    const _Float16* p = row + 32 * t + 16 * half;
    return frag_cat8(*(const v8h*)p, *(const v8h*)(p + 8));
}

static __device__ __forceinline__ v16h load_a_f32(const float* row, int t, int half) {
    const float* p = row + 32 * t + 16 * half;
    v8h a, b;
#pragma unroll
    for (int i = 0; i < 8; ++i) { a[i] = (_Float16)p[i]; b[i] = (_Float16)p[i + 8]; }
    return frag_cat8(a, b);
}

// B-matrix (32x16 f16) fragment: lane holds column (lane&15); halfword e -> K = 16*half + e
// => 16 sequential halfwords; caller passes pointer to K offset 16*half within the row.
static __device__ __forceinline__ v16h load_b_seq(const _Float16* p) {
    return frag_cat8(*(const v8h*)p, *(const v8h*)(p + 8));
}

// Async global -> LDS copy, 16 bytes per lane, tracked by ASYNCcnt.
static __device__ __forceinline__ void async_b128(unsigned ldsoff, const _Float16* gaddr) {
    asm volatile("global_load_async_to_lds_b128 %0, %1, off"
                 :: "v"(ldsoff), "v"(gaddr) : "memory");
}
#define ASYNC_WAIT0() asm volatile("s_wait_asynccnt 0" ::: "memory")

static __device__ __forceinline__ unsigned lds_off(const void* p) {
    // generic pointer to LDS: low 32 bits are the LDS offset (aperture in high bits)
    return (unsigned)(unsigned long long)p;
}

// --------------------------------------------------------------------------
// f32 [B,C,hw] channel-major  ->  f16 [B,hw,C] pixel-major
__global__ void __launch_bounds__(256)
t_kernel(const float* __restrict__ in, _Float16* __restrict__ out, int hw, int total) {
    int idx = blockIdx.x * 256 + threadIdx.x;
    if (idx >= total) return;
    int s = idx % hw;
    int rest = idx / hw;          // b*128 + c
    int c = rest & 127;
    int b = rest >> 7;
    out[((size_t)b * hw + s) * C_CH + c] = (_Float16)in[idx];
}

// --------------------------------------------------------------------------
// Fused 1x1 conv + BN + LeakyReLU.  X: f16 pixel-major [hw,C].
// mode 0: Out f16 pixel-major [hw,C];  mode 1: Out f16 channel-major [C,hw].
__global__ void __launch_bounds__(128)
proj_kernel(const _Float16* __restrict__ X, const float* __restrict__ Wm,
            const float* __restrict__ g, const float* __restrict__ bvec,
            _Float16* __restrict__ Out, int hw, int mode) {
    const int lane = threadIdx.x & 31;
    const int wave = threadIdx.x >> 5;
    const int half = lane >> 4;
    const int ln   = lane & 15;
    const int s0   = blockIdx.x * 64;
    const int m0   = blockIdx.y * 64 + wave * 16;
    const int b    = blockIdx.z;

    X   += (size_t)b * hw * C_CH;
    Out += (size_t)b * hw * C_CH;

    const float* wrow = Wm + (size_t)(m0 + ln) * C_CH;
    v16h A[4];
#pragma unroll
    for (int t = 0; t < 4; ++t) A[t] = load_a_f32(wrow, t, half);

    float sc[8], bs[8];
#pragma unroll
    for (int r = 0; r < 8; ++r) {
        int o = m0 + 8 * half + r;
        sc[r] = g[o] * BN_SC;
        bs[r] = bvec[o];
    }

    v8f acc[4];
#pragma unroll
    for (int n = 0; n < 4; ++n) {
        v8f c = {};
        const _Float16* xrow = X + (size_t)(s0 + 16 * n + ln) * C_CH;
#pragma unroll
        for (int t = 0; t < 4; ++t) {
            v16h Bf = load_b_seq(xrow + 32 * t + 16 * half);
            c = wmma16(A[t], Bf, c);
        }
        acc[n] = c;
    }

    if (mode == 0) {
#pragma unroll
        for (int n = 0; n < 4; ++n) {
            int srow = s0 + 16 * n + ln;
            v8h y;
#pragma unroll
            for (int r = 0; r < 8; ++r) {
                float v = acc[n][r] * sc[r] + bs[r];
                v = v > 0.f ? v : 0.1f * v;
                y[r] = (_Float16)v;
            }
            *(v8h*)(Out + (size_t)srow * C_CH + m0 + 8 * half) = y;
        }
    } else {
#pragma unroll
        for (int r = 0; r < 8; ++r) {
            int o = m0 + 8 * half + r;
#pragma unroll
            for (int n = 0; n < 4; ++n) {
                int srow = s0 + 16 * n + ln;
                float v = acc[n][r] * sc[r] + bs[r];
                v = v > 0.f ? v : 0.1f * v;
                Out[(size_t)o * hw + srow] = (_Float16)v;
            }
        }
    }
}

// --------------------------------------------------------------------------
// Flash attention: Q,K f16 pixel-major [hw,C]; V f16 channel-major [C,hw].
// Ctx f16 pixel-major [hw,C].  One wave = 16 query rows; WG = 64 queries.
// K and V tiles staged in LDS via global_load_async_to_lds_b128 (ASYNCcnt).
__global__ void __launch_bounds__(128)
attn_kernel(const _Float16* __restrict__ Q, const _Float16* __restrict__ K,
            const _Float16* __restrict__ Vcm, _Float16* __restrict__ Ctx, int hw) {
    __shared__ __align__(32) _Float16 klds[64][136];     // K tile [key][c], padded
    __shared__ __align__(32) _Float16 vlds[128][72];     // V tile [c][key], padded
    __shared__ __align__(32) _Float16 plds[4][16][72];   // per-wave P scratch

    const int tid  = threadIdx.x;
    const int lane = tid & 31;
    const int wave = tid >> 5;
    const int half = lane >> 4;
    const int ln   = lane & 15;
    const int b    = blockIdx.y;
    const int q0   = blockIdx.x * 64 + wave * 16;

    Q   += (size_t)b * hw * C_CH;
    K   += (size_t)b * hw * C_CH;
    Vcm += (size_t)b * C_CH * hw;
    Ctx += (size_t)b * hw * C_CH;

    const _Float16* qrow = Q + (size_t)(q0 + ln) * C_CH;
    v16h qa[4];
#pragma unroll
    for (int t = 0; t < 4; ++t) qa[t] = load_a_f16(qrow, t, half);

    v8f Oacc[8];
    float rm[8], rs[8];
#pragma unroll
    for (int nt = 0; nt < 8; ++nt) Oacc[nt] = (v8f){};
#pragma unroll
    for (int r = 0; r < 8; ++r) { rm[r] = -1e30f; rs[r] = 0.f; }

    const float scale = 0.08838834764831845f;   // 1/sqrt(128)

    // per-thread staging assignments (128 threads)
    const int krow = tid >> 1, kseg = tid & 1;   // K tile: 64 rows x 256B

    for (int kb = 0; kb < hw; kb += 64) {
        __syncthreads();   // all waves done reading previous tile
        {   // async-stage 64x128 K tile and 128x64 V tile (16 b128s / thread)
            const _Float16* gk = K + (size_t)(kb + krow) * C_CH + 64 * kseg;
            _Float16* dk = &klds[krow][64 * kseg];
#pragma unroll
            for (int i = 0; i < 8; ++i)
                async_b128(lds_off(dk + 8 * i), gk + 8 * i);

            const _Float16* gv = Vcm + (size_t)tid * hw + kb;
            _Float16* dv = &vlds[tid][0];
#pragma unroll
            for (int i = 0; i < 8; ++i)
                async_b128(lds_off(dv + 8 * i), gv + 8 * i);
        }
        ASYNC_WAIT0();
        __syncthreads();   // tile visible to all waves

        // S = Q * K^T  (16 x 64 per wave)
        v8f S[4];
#pragma unroll
        for (int n = 0; n < 4; ++n) {
            v8f c = {};
            const _Float16* krow_p = &klds[16 * n + ln][0];
#pragma unroll
            for (int t = 0; t < 4; ++t) {
                v16h Bf = load_b_seq(krow_p + 32 * t + 16 * half);
                c = wmma16(qa[t], Bf, c);
            }
            S[n] = c;
        }

        // online softmax (rows split: half0 -> rows 0..7, half1 -> rows 8..15)
#pragma unroll
        for (int n = 0; n < 4; ++n) S[n] *= scale;

#pragma unroll
        for (int r = 0; r < 8; ++r) {
            float mx = fmaxf(fmaxf(S[0][r], S[1][r]), fmaxf(S[2][r], S[3][r]));
#pragma unroll
            for (int d = 1; d < 16; d <<= 1)
                mx = fmaxf(mx, __shfl_xor(mx, d, 32));
            float nm = fmaxf(rm[r], mx);
            float al = __expf(rm[r] - nm);
            rm[r] = nm;
            float ps = 0.f;
#pragma unroll
            for (int n = 0; n < 4; ++n) {
                float p = __expf(S[n][r] - nm);
                S[n][r] = p;
                ps += p;
            }
#pragma unroll
            for (int d = 1; d < 16; d <<= 1)
                ps += __shfl_xor(ps, d, 32);
            rs[r] = rs[r] * al + ps;
#pragma unroll
            for (int nt = 0; nt < 8; ++nt) Oacc[nt][r] *= al;
        }

        // C/D layout -> A layout via per-wave LDS scratch
#pragma unroll
        for (int r = 0; r < 8; ++r)
#pragma unroll
            for (int n = 0; n < 4; ++n)
                plds[wave][r + 8 * half][16 * n + ln] = (_Float16)S[n][r];

        v16h pa[2];
#pragma unroll
        for (int t = 0; t < 2; ++t)
            pa[t] = load_a_f16(&plds[wave][ln][0], t, half);

        // O += P * V   (V tile in LDS, channel-major rows)
#pragma unroll
        for (int nt = 0; nt < 8; ++nt) {
            const _Float16* vrow = &vlds[16 * nt + ln][0];
#pragma unroll
            for (int t = 0; t < 2; ++t) {
                v16h Bf = load_b_seq(vrow + 32 * t + 16 * half);
                Oacc[nt] = wmma16(pa[t], Bf, Oacc[nt]);
            }
        }
    }

    // normalize + store ctx (pixel-major f16)
#pragma unroll
    for (int r = 0; r < 8; ++r) {
        float inv = 1.0f / rs[r];
        int qr = q0 + r + 8 * half;
#pragma unroll
        for (int nt = 0; nt < 8; ++nt)
            Ctx[(size_t)qr * C_CH + 16 * nt + ln] = (_Float16)(Oacc[nt][r] * inv);
    }
}

// --------------------------------------------------------------------------
// Out-projection + BN + LeakyReLU + residual. Output f32 [B,C,hw].
__global__ void __launch_bounds__(128)
outproj_kernel(const _Float16* __restrict__ X, const float* __restrict__ Wm,
               const float* __restrict__ g, const float* __restrict__ bvec,
               const float* __restrict__ resid, float* __restrict__ Out, int hw) {
    const int lane = threadIdx.x & 31;
    const int wave = threadIdx.x >> 5;
    const int half = lane >> 4;
    const int ln   = lane & 15;
    const int s0   = blockIdx.x * 64;
    const int m0   = blockIdx.y * 64 + wave * 16;
    const int b    = blockIdx.z;

    X     += (size_t)b * hw * C_CH;
    resid += (size_t)b * C_CH * hw;
    Out   += (size_t)b * C_CH * hw;

    const float* wrow = Wm + (size_t)(m0 + ln) * C_CH;
    v16h A[4];
#pragma unroll
    for (int t = 0; t < 4; ++t) A[t] = load_a_f32(wrow, t, half);

    float sc[8], bs[8];
#pragma unroll
    for (int r = 0; r < 8; ++r) {
        int o = m0 + 8 * half + r;
        sc[r] = g[o] * BN_SC;
        bs[r] = bvec[o];
    }

    v8f acc[4];
#pragma unroll
    for (int n = 0; n < 4; ++n) {
        v8f c = {};
        const _Float16* xrow = X + (size_t)(s0 + 16 * n + ln) * C_CH;
#pragma unroll
        for (int t = 0; t < 4; ++t) {
            v16h Bf = load_b_seq(xrow + 32 * t + 16 * half);
            c = wmma16(A[t], Bf, c);
        }
        acc[n] = c;
    }

#pragma unroll
    for (int r = 0; r < 8; ++r) {
        int o = m0 + 8 * half + r;
#pragma unroll
        for (int n = 0; n < 4; ++n) {
            int srow = s0 + 16 * n + ln;
            float v = acc[n][r] * sc[r] + bs[r];
            v = v > 0.f ? v : 0.1f * v;
            size_t idx = (size_t)o * hw + srow;
            Out[idx] = v + resid[idx];
        }
    }
}

// --------------------------------------------------------------------------
extern "C" void kernel_launch(void* const* d_in, const int* in_sizes, int n_in,
                              void* d_out, int out_size, void* d_ws, size_t ws_size,
                              hipStream_t stream) {
    (void)in_sizes; (void)n_in; (void)out_size; (void)ws_size;
    const int B = 2, H = 64, Wd = 96;
    const int hw = H * Wd;                       // 6144

    const float* query = (const float*)d_in[0];
    const float* key   = (const float*)d_in[1];
    const float* q_w1 = (const float*)d_in[2];
    const float* q_g1 = (const float*)d_in[3];
    const float* q_b1 = (const float*)d_in[4];
    const float* q_w2 = (const float*)d_in[5];
    const float* q_g2 = (const float*)d_in[6];
    const float* q_b2 = (const float*)d_in[7];
    const float* k_w1 = (const float*)d_in[8];
    const float* k_g1 = (const float*)d_in[9];
    const float* k_b1 = (const float*)d_in[10];
    const float* k_w2 = (const float*)d_in[11];
    const float* k_g2 = (const float*)d_in[12];
    const float* k_b2 = (const float*)d_in[13];
    const float* v_w  = (const float*)d_in[14];
    const float* v_g  = (const float*)d_in[15];
    const float* v_b  = (const float*)d_in[16];
    const float* o_w  = (const float*)d_in[17];
    const float* o_g  = (const float*)d_in[18];
    const float* o_b  = (const float*)d_in[19];

    const size_t nElem = (size_t)B * C_CH * hw;  // 1,572,864

    char* ws = (char*)d_ws;
    size_t off = 0;
    auto alloc16 = [&](size_t elems) -> _Float16* {
        _Float16* p = (_Float16*)(ws + off);
        off += ((elems * 2 + 255) / 256) * 256;
        return p;
    };
    _Float16* xqh = alloc16(nElem);   // query, f16 pixel-major
    _Float16* xkh = alloc16(nElem);   // key,   f16 pixel-major
    _Float16* t1  = alloc16(nElem);   // q layer1 out
    _Float16* t2  = alloc16(nElem);   // k layer1 out
    _Float16* qh  = alloc16(nElem);   // q, pixel-major
    _Float16* kh  = alloc16(nElem);   // k, pixel-major
    _Float16* vcm = alloc16(nElem);   // v, channel-major
    _Float16* ctx = alloc16(nElem);   // attention out, pixel-major

    const int total = (int)nElem;
    t_kernel<<<(total + 255) / 256, 256, 0, stream>>>(query, xqh, hw, total);
    t_kernel<<<(total + 255) / 256, 256, 0, stream>>>(key,   xkh, hw, total);

    dim3 gp(hw / 64, C_CH / 64, B);
    proj_kernel<<<gp, 128, 0, stream>>>(xqh, q_w1, q_g1, q_b1, t1,  hw, 0);
    proj_kernel<<<gp, 128, 0, stream>>>(t1,  q_w2, q_g2, q_b2, qh,  hw, 0);
    proj_kernel<<<gp, 128, 0, stream>>>(xkh, k_w1, k_g1, k_b1, t2,  hw, 0);
    proj_kernel<<<gp, 128, 0, stream>>>(t2,  k_w2, k_g2, k_b2, kh,  hw, 0);
    proj_kernel<<<gp, 128, 0, stream>>>(xkh, v_w,  v_g,  v_b,  vcm, hw, 1);

    attn_kernel<<<dim3(hw / 64, B), 128, 0, stream>>>(qh, kh, vcm, ctx, hw);

    outproj_kernel<<<gp, 128, 0, stream>>>(ctx, o_w, o_g, o_b, query, (float*)d_out, hw);
}
